// DeformConvWrapper_20478404067805
// MI455X (gfx1250) — matrix operand order repaired and checked by reading, not compiled, not run on hardware
//
#include <hip/hip_runtime.h>
#include <math.h>

// ---------------- problem constants ----------------
#define Bq    4
#define Cq    64
#define Hq    160
#define Wq    160
#define OUTq  64
#define K2q   9
#define HWq   (Hq * Wq)          // 25600 pixels per batch image
#define PPB   HWq
#define CK    (Cq * K2q)         // 576 reduction length (ck = c*9 + tap)
#define KSTEPS 18                // 576 / 32
#define GROUPS_PER_B (PPB / 16)  // 1600 sixteen-pixel groups per batch
#define WAVES_PER_BLOCK 8
#define AF_DW  (4 * KSTEPS * 32 * 8)   // main GEMM A fragments: 18432 dwords
#define CAF_DW (2 * KSTEPS * 32 * 8)   // conv A fragments: 9216 dwords

typedef __attribute__((ext_vector_type(16))) __bf16 v16bf;
typedef __attribute__((ext_vector_type(8)))  float  v8f;
typedef unsigned int u32x4 __attribute__((ext_vector_type(4)));
typedef int          i32x4 __attribute__((ext_vector_type(4)));
typedef int          i32x8 __attribute__((ext_vector_type(8)));

// ---- A-matrix (16x32 bf16) per-lane K mapping used here: contiguous, k = 16*hi + 2v ----
__device__ __forceinline__ int kmapA(int v, int hi) {
  return 2 * v + (hi ? 16 : 0);
}

__device__ __forceinline__ unsigned pack2bf(float a, float b) {  // RNE f32->bf16 x2
  unsigned ua = __float_as_uint(a), ub = __float_as_uint(b);
  ua += 0x7FFFu + ((ua >> 16) & 1u);
  ub += 0x7FFFu + ((ub >> 16) & 1u);
  return (ua >> 16) | (ub & 0xFFFF0000u);
}

__device__ __forceinline__ int iclamp(int v, int lo, int hi) {
  return v < lo ? lo : (v > hi ? hi : v);
}

// ============================================================================
// Kernel 0: pre-swizzle weights into bf16 WMMA A fragments.
//  - main:  dcn_w   (64 x 576)  -> 4 M-tiles  -> AF
//  - conv:  offset_w(18 x 576) ++ mask_w(9 x 576) ++ 5 zero rows -> 2 M-tiles -> CAF
// ============================================================================
__device__ __forceinline__ float conv_w_row(int row, int K,
                                            const float* __restrict__ ow,
                                            const float* __restrict__ mw) {
  if (row < 18) return ow[row * CK + K];
  if (row < 27) return mw[(row - 18) * CK + K];
  return 0.0f;
}

__global__ void prep_weights_kernel(const float* __restrict__ dcn_w,
                                    const float* __restrict__ offset_w,
                                    const float* __restrict__ mask_w,
                                    unsigned* __restrict__ AF,
                                    unsigned* __restrict__ CAF) {
  int tid = blockIdx.x * blockDim.x + threadIdx.x;
  if (tid < 4 * KSTEPS * 32) {
    int lane = tid & 31, kstep = (tid >> 5) % KSTEPS, mt = tid / (KSTEPS * 32);
    int m = mt * 16 + (lane & 15), hi = lane >> 4;
    unsigned* dst = AF + (size_t)((mt * KSTEPS + kstep) * 32 + lane) * 8;
#pragma unroll
    for (int v = 0; v < 8; ++v) {
      int K = kstep * 32 + kmapA(v, hi);
      dst[v] = pack2bf(dcn_w[m * CK + K], dcn_w[m * CK + K + 1]);
    }
  } else if (tid < 6 * KSTEPS * 32) {
    int t2 = tid - 4 * KSTEPS * 32;
    int lane = t2 & 31, kstep = (t2 >> 5) % KSTEPS, mt = t2 / (KSTEPS * 32);
    int m = mt * 16 + (lane & 15), hi = lane >> 4;
    unsigned* dst = CAF + (size_t)((mt * KSTEPS + kstep) * 32 + lane) * 8;
#pragma unroll
    for (int v = 0; v < 8; ++v) {
      int K = kstep * 32 + kmapA(v, hi);
      dst[v] = pack2bf(conv_w_row(m, K, offset_w, mask_w),
                       conv_w_row(m, K + 1, offset_w, mask_w));
    }
  }
}

// ============================================================================
// Fused kernel, per wave of 16 pixels:
//   phase 0: (wave 0) issue TDM load of all main A fragments into LDS
//   phase C: build conv tap table {clamped offset, validity}
//   phase 1: offset/mask 3x3 conv via WMMA (2 M-tiles x 18 K-steps) -> raw LDS
//   phase 2: sigmoid + bilinear corner/weight table (mask & validity folded)
//   phase 3: main GEMM via WMMA (4 M-tiles x 18 K-steps), A from LDS
// Each lane's 16 B-elements per K-step are CONSECUTIVE in K, so (c, tap) are
// tracked incrementally (no integer division anywhere in the hot loops).
// ============================================================================
struct Samp { int i00, i01, i10, i11; float w00, w01, w10, w11; };
struct CTap { int ofs; float val; };

__global__ void __launch_bounds__(256)
dcn_fused_kernel(const float* __restrict__ x,
                 const float* __restrict__ offset_b,
                 const float* __restrict__ mask_b,
                 const unsigned* __restrict__ AF,
                 const unsigned* __restrict__ CAF,
                 float* __restrict__ out) {
  __shared__ __align__(16) unsigned lds_a[AF_DW];          // 73,728 B
  __shared__ Samp  table[WAVES_PER_BLOCK][K2q * 16];       // 36,864 B
  __shared__ float raw[WAVES_PER_BLOCK][16][28];           // 14,336 B
  __shared__ CTap  ctab[WAVES_PER_BLOCK][K2q * 16];        //  9,216 B

  int lane = threadIdx.x & 31;
  int wave = threadIdx.x >> 5;
  int group = blockIdx.x * WAVES_PER_BLOCK + wave;         // 0..6399
  int b   = group / GROUPS_PER_B;
  int pp0 = (group % GROUPS_PER_B) * 16;
  int i0 = pp0 / Wq, j0 = pp0 % Wq;                        // group row / col base
  int n  = lane & 15;                                      // pixel column (N)
  int hi = lane >> 4;
  const float* xb = x + (size_t)b * Cq * HWq;

  // ---------------- phase 0: issue TDM async copy (wait deferred) ----------
  if (wave == 0) {
    unsigned long long ga = (unsigned long long)AF;
    unsigned lofs = (unsigned)(size_t)&lds_a[0];           // low 32 bits = LDS offset
    u32x4 g0;
    g0.x = 1u;                                             // count=1, user mode
    g0.y = lofs;                                           // lds_addr
    g0.z = (unsigned)(ga & 0xFFFFFFFFu);                   // global_addr[31:0]
    g0.w = (unsigned)((ga >> 32) & 0x01FFFFFFu) | (2u << 30);  // addr[56:32] | type=2
    i32x8 g1;
    g1[0] = (2 << 16);                                     // data_size = 4B
    g1[1] = (int)((AF_DW & 0xFFFF) << 16);                 // tensor_dim0[15:0]
    g1[2] = (int)((AF_DW >> 16) | (1 << 16));              // dim0[31:16] | tensor_dim1=1
    g1[3] = (int)((AF_DW & 0xFFFF) << 16);                 // tile_dim0 = 18432
    g1[4] = 1;                                             // tile_dim1 = 1
    g1[5] = AF_DW;                                         // tensor_dim0_stride
    g1[6] = 0; g1[7] = 0;
    i32x4 gz = {0, 0, 0, 0};
#if defined(__clang_major__) && (__clang_major__ >= 23)
    i32x8 gz8 = {0, 0, 0, 0, 0, 0, 0, 0};
    __builtin_amdgcn_tensor_load_to_lds(g0, g1, gz, gz, gz8, 0);
#else
    __builtin_amdgcn_tensor_load_to_lds(g0, g1, gz, gz, 0);
#endif
  }

  // ---------------- phase C: conv tap table (branchless gather helper) -----
  for (int t = lane; t < K2q * 16; t += 32) {
    int k = t >> 4, nn = t & 15;
    int kh = k / 3, kw = k % 3;                            // k < 9: cheap
    int yi = i0 - 1 + kh;
    int xj = j0 + nn - 1 + kw;
    CTap e;
    e.ofs = iclamp(yi, 0, Hq - 1) * Wq + iclamp(xj, 0, Wq - 1);
    e.val = ((unsigned)yi < (unsigned)Hq && (unsigned)xj < (unsigned)Wq) ? 1.0f : 0.0f;
    ctab[wave][t] = e;
  }
  __syncthreads();

  // ---------------- phase 1: offset/mask conv via WMMA ---------------------
  {
    v8f cacc0 = {}, cacc1 = {};
    int tapOfs = hi ? 7 * 16 : 0;      // K0 = 16*hi: hi=1 -> c=1, tap=7
    int cofs   = hi ? HWq : 0;
    for (int kstep = 0; kstep < KSTEPS; ++kstep) {
      float fv[16];
#pragma unroll
      for (int q = 0; q < 16; ++q) {
        CTap e = ctab[wave][tapOfs + n];
        fv[q] = xb[cofs + e.ofs] * e.val;
        tapOfs += 16;
        if (tapOfs == K2q * 16) { tapOfs = 0; cofs += HWq; }
      }
      union { unsigned u[8]; v16bf v; } cb;
#pragma unroll
      for (int v = 0; v < 8; ++v) cb.u[v] = pack2bf(fv[2 * v], fv[2 * v + 1]);

      const unsigned* cab = CAF + (size_t)(kstep * 32 + lane) * 8;
      __builtin_prefetch(cab + 32 * 8, 0, 0);
      union { unsigned u[8]; v16bf v; } ca0, ca1;
#pragma unroll
      for (int v = 0; v < 8; ++v) {
        ca0.u[v] = cab[v];
        ca1.u[v] = cab[KSTEPS * 32 * 8 + v];
      }
      cacc0 = __builtin_amdgcn_wmma_f32_16x16x32_bf16(false, ca0.v, false, cb.v, (short)0, cacc0, false, false);
      cacc1 = __builtin_amdgcn_wmma_f32_16x16x32_bf16(false, ca1.v, false, cb.v, (short)0, cacc1, false, false);

      // advance K by the other half-wave's 16: c += 1, tap += 7
      cofs += HWq;
      tapOfs += 7 * 16;
      if (tapOfs >= K2q * 16) { tapOfs -= K2q * 16; cofs += HWq; }
    }
    // scatter raw conv channels (+bias) to LDS: ch = mt*16 + r + 8*hi
#pragma unroll
    for (int r = 0; r < 8; ++r) {
      int ch0 = r + 8 * hi;                                // 0..15 (offset y/x)
      raw[wave][n][ch0] = cacc0[r] + offset_b[ch0];
      int ch1 = 16 + r + 8 * hi;                           // 16..31
      if (ch1 < 27) {
        float bias = (ch1 < 18) ? offset_b[ch1] : mask_b[ch1 - 18];
        raw[wave][n][ch1] = cacc1[r] + bias;
      }
    }
  }
  __syncthreads();   // raw[] ready

  // ---------------- phase 2: bilinear corner/weight table ------------------
  for (int t = lane; t < K2q * 16; t += 32) {
    int k = t >> 4, nn = t & 15;
    int pp = pp0 + nn;
    int ii = pp / Wq, jj = pp % Wq;
    float offy = raw[wave][nn][2 * k];
    float offx = raw[wave][nn][2 * k + 1];
    float mv   = 1.0f / (1.0f + __expf(-raw[wave][nn][18 + k]));
    float py = offy + (float)(ii - 1) + (float)(k / 3);
    float px = offx + (float)(jj - 1) + (float)(k % 3);
    float y0f = floorf(py), x0f = floorf(px);
    float wy1 = py - y0f, wx1 = px - x0f;
    float wy0 = 1.0f - wy1, wx0 = 1.0f - wx1;
    int y0 = (int)y0f, x0 = (int)x0f, y1 = y0 + 1, x1 = x0 + 1;
    float vy0 = ((unsigned)y0 < (unsigned)Hq) ? 1.f : 0.f;
    float vy1 = ((unsigned)y1 < (unsigned)Hq) ? 1.f : 0.f;
    float vx0 = ((unsigned)x0 < (unsigned)Wq) ? 1.f : 0.f;
    float vx1 = ((unsigned)x1 < (unsigned)Wq) ? 1.f : 0.f;
    int cy0 = iclamp(y0, 0, Hq - 1), cy1 = iclamp(y1, 0, Hq - 1);
    int cx0 = iclamp(x0, 0, Wq - 1), cx1 = iclamp(x1, 0, Wq - 1);
    Samp s;
    s.i00 = cy0 * Wq + cx0; s.i01 = cy0 * Wq + cx1;
    s.i10 = cy1 * Wq + cx0; s.i11 = cy1 * Wq + cx1;
    s.w00 = wy0 * wx0 * vy0 * vx0 * mv;
    s.w01 = wy0 * wx1 * vy0 * vx1 * mv;
    s.w10 = wy1 * wx0 * vy1 * vx0 * mv;
    s.w11 = wy1 * wx1 * vy1 * vx1 * mv;
    table[wave][t] = s;
  }
  if (wave == 0) __builtin_amdgcn_s_wait_tensorcnt(0);  // lds_a ready before barrier
  __syncthreads();

  // ---------------- phase 3: main GEMM via WMMA (A from LDS) ---------------
  v8f acc0 = {}, acc1 = {}, acc2 = {}, acc3 = {};
  {
    int tapOfs = hi ? 7 * 16 : 0;
    int cofs   = hi ? HWq : 0;
    for (int kstep = 0; kstep < KSTEPS; ++kstep) {
      float fv[16];
#pragma unroll
      for (int q = 0; q < 16; ++q) {
        Samp s = table[wave][tapOfs + n];
        const float* xc = xb + cofs;
        fv[q] = s.w00 * xc[s.i00] + s.w01 * xc[s.i01] +
                s.w10 * xc[s.i10] + s.w11 * xc[s.i11];
        tapOfs += 16;
        if (tapOfs == K2q * 16) { tapOfs = 0; cofs += HWq; }
      }
      union { unsigned u[8]; v16bf v; } bf;
#pragma unroll
      for (int v = 0; v < 8; ++v) bf.u[v] = pack2bf(fv[2 * v], fv[2 * v + 1]);

      int lb = (kstep * 32 + lane) * 8;
      const uint4* ap = (const uint4*)&lds_a[lb];
      union { uint4 q[2]; v16bf v; } a0, a1, a2, a3;
      a0.q[0] = ap[0];                      a0.q[1] = ap[1];
      a1.q[0] = ap[KSTEPS * 32 * 2];        a1.q[1] = ap[KSTEPS * 32 * 2 + 1];
      a2.q[0] = ap[2 * KSTEPS * 32 * 2];    a2.q[1] = ap[2 * KSTEPS * 32 * 2 + 1];
      a3.q[0] = ap[3 * KSTEPS * 32 * 2];    a3.q[1] = ap[3 * KSTEPS * 32 * 2 + 1];
      acc0 = __builtin_amdgcn_wmma_f32_16x16x32_bf16(false, a0.v, false, bf.v, (short)0, acc0, false, false);
      acc1 = __builtin_amdgcn_wmma_f32_16x16x32_bf16(false, a1.v, false, bf.v, (short)0, acc1, false, false);
      acc2 = __builtin_amdgcn_wmma_f32_16x16x32_bf16(false, a2.v, false, bf.v, (short)0, acc2, false, false);
      acc3 = __builtin_amdgcn_wmma_f32_16x16x32_bf16(false, a3.v, false, bf.v, (short)0, acc3, false, false);

      cofs += HWq;
      tapOfs += 7 * 16;
      if (tapOfs >= K2q * 16) { tapOfs -= K2q * 16; cofs += HWq; }
    }
  }

  // ---------------- epilogue: C/D layout -> out[b][o][pp] ------------------
  int pp = pp0 + n;
  float* ob = out + (size_t)b * OUTq * HWq + pp;
#pragma unroll
  for (int r = 0; r < 8; ++r) {
    int row = r + hi * 8;
    ob[(size_t)(0 * 16 + row) * HWq] = acc0[r];
    ob[(size_t)(1 * 16 + row) * HWq] = acc1[r];
    ob[(size_t)(2 * 16 + row) * HWq] = acc2[r];
    ob[(size_t)(3 * 16 + row) * HWq] = acc3[r];
  }
}

// ============================================================================
extern "C" void kernel_launch(void* const* d_in, const int* in_sizes, int n_in,
                              void* d_out, int out_size, void* d_ws, size_t ws_size,
                              hipStream_t stream) {
  const float* x        = (const float*)d_in[0];
  const float* offset_w = (const float*)d_in[1];
  const float* offset_b = (const float*)d_in[2];
  const float* mask_w   = (const float*)d_in[3];
  const float* mask_b   = (const float*)d_in[4];
  const float* dcn_w    = (const float*)d_in[5];
  float* out = (float*)d_out;

  unsigned* AF  = (unsigned*)d_ws;          // 18432 dwords
  unsigned* CAF = AF + AF_DW;               //  9216 dwords

  prep_weights_kernel<<<dim3((6 * KSTEPS * 32 + 255) / 256), dim3(256), 0, stream>>>(
      dcn_w, offset_w, mask_w, AF, CAF);
  dcn_fused_kernel<<<dim3((Bq * GROUPS_PER_B) / WAVES_PER_BLOCK), dim3(256), 0, stream>>>(
      x, offset_b, mask_b, AF, CAF, out);
}
